// GPT_77773267796127
// MI455X (gfx1250) — compile-verified, matrix operand order checked by reading, not run on hardware
//
#include <hip/hip_runtime.h>

// ---------------------------------------------------------------------------
// GPT block forward for MI455X (gfx1250, wave32, WMMA, async global->LDS).
// All matmuls: bf16 operands + f32 accumulation (V_WMMA_F32_16X16X32_BF16).
// Double-buffered async pipeline: issue tile k+1 copies, s_wait_asynccnt N,
// barrier, ds_load_b128 fragments, v_wmma. No conversion VALU in hot loops.
// ---------------------------------------------------------------------------

typedef __attribute__((ext_vector_type(16))) __bf16 v16bf;
typedef __attribute__((ext_vector_type(8)))  __bf16 v8bf;
typedef __attribute__((ext_vector_type(8)))  float  v8f;

#define D_MODEL   1024
#define N_HEAD    16
#define HEAD_DIM  64
#define SEQ_L     2048
#define BATCH     2
#define M_TOT     (BATCH * SEQ_L)   // 4096
#define IN_DIM    512
#define MLP_HID   4096

// ------------------------- CDNA5 async copy helpers ------------------------
#if defined(__has_builtin)
#  if __has_builtin(__builtin_amdgcn_global_load_async_to_lds_b128)
#    define ATH_ASYNC 1
#  endif
#  if __has_builtin(__builtin_amdgcn_s_wait_asynccnt)
#    define ATH_WAITA 1
#  endif
#endif
#ifndef ATH_ASYNC
#define ATH_ASYNC 0
#endif
#ifndef ATH_WAITA
#define ATH_WAITA 0
#endif

// ROCm builtin signature uses GCC-vector int4 pointees:
//   void __builtin_amdgcn_global_load_async_to_lds_b128(v4i AS1* src,
//                                                       v4i AS3* dst,
//                                                       imm offset, imm cpol)
typedef int v4i_gcc __attribute__((vector_size(16)));
typedef __attribute__((address_space(1))) v4i_gcc* as1_v4i_t;   // 64-bit ptr
typedef __attribute__((address_space(3))) v4i_gcc* as3_v4i_t;   // 32-bit ptr

// 16-byte async copy global -> LDS (GLOBAL_LOAD_ASYNC_TO_LDS_B128).
// AS3 pointer == low 32 bits of the flat LDS address (LDS offset).
__device__ __forceinline__ void cp_async16(__bf16* lds_dst, const __bf16* gsrc) {
#if ATH_ASYNC
    as3_v4i_t l = __builtin_bit_cast(as3_v4i_t, (unsigned)(unsigned long long)lds_dst);
    as1_v4i_t g = __builtin_bit_cast(as1_v4i_t, (unsigned long long)gsrc);
    __builtin_amdgcn_global_load_async_to_lds_b128(g, l, 0, 0);
#else
    *(v8bf*)lds_dst = *(const v8bf*)gsrc;
#endif
}

// Wait until at most N async transfers remain in flight (in-order completion
// => everything older than the newest N has landed in LDS).
template<int N>
__device__ __forceinline__ void wait_async_n() {
#if ATH_ASYNC
#  if ATH_WAITA
    __builtin_amdgcn_s_wait_asynccnt(N);
#  else
    asm volatile("s_wait_asynccnt %0" :: "i"(N) : "memory");
#  endif
#endif
}

// ------------------------------- bf16 utils --------------------------------
__device__ __forceinline__ __bf16 f2bf(float f) {
    unsigned u = __builtin_bit_cast(unsigned, f);
    u += 0x7FFFu + ((u >> 16) & 1u);           // round-to-nearest-even
    unsigned short s = (unsigned short)(u >> 16);
    return __builtin_bit_cast(__bf16, s);
}

__device__ __forceinline__ v8f wmma_bf16(v16bf a, v16bf b, v8f c) {
    return __builtin_amdgcn_wmma_f32_16x16x32_bf16(
        false, a, false, b, (short)0, c, false, false);
}

// WMMA 16-bit operand fragment from LDS: lane's K run lives at
// {kb..kb+7, kb+16..kb+23}, kb = 8*(lane>=16). Two ds_load_b128.
__device__ __forceinline__ v16bf load_frag(const __bf16* base, int kb) {
    v8bf lo = *(const v8bf*)(base + kb);
    v8bf hi = *(const v8bf*)(base + kb + 16);
    v16bf r;
#pragma unroll
    for (int j = 0; j < 8; ++j) { r[j] = lo[j]; r[8 + j] = hi[j]; }
    return r;
}

// ---------------------------------------------------------------------------
// Prepass: fp32 -> bf16 elementwise (activations).
// ---------------------------------------------------------------------------
__global__ __launch_bounds__(256)
void cvt_bf16_kernel(const float* __restrict__ x, __bf16* __restrict__ y, int n) {
    int i = blockIdx.x * 256 + threadIdx.x;
    if (i < n) y[i] = f2bf(x[i]);
}

// Prepass: weight transpose+convert  W[K][N] fp32 -> Wt[N][K] bf16.
__global__ __launch_bounds__(256)
void wtrans_kernel(const float* __restrict__ w, __bf16* __restrict__ wt, int K, int N) {
    __shared__ float t[32][33];
    const int k0 = blockIdx.y * 32, n0 = blockIdx.x * 32;
    const int tx = threadIdx.x & 31, ty = threadIdx.x >> 5;     // 32 x 8
#pragma unroll
    for (int i = 0; i < 32; i += 8) t[ty + i][tx] = w[(size_t)(k0 + ty + i) * N + n0 + tx];
    __syncthreads();
#pragma unroll
    for (int i = 0; i < 32; i += 8) wt[(size_t)(n0 + ty + i) * K + k0 + tx] = f2bf(t[tx][ty + i]);
}

// Prepass: V [B*L][D_MODEL] fp32 -> Vt [B*H][64][L] bf16 (per-head transpose).
__global__ __launch_bounds__(256)
void vtrans_kernel(const float* __restrict__ v, __bf16* __restrict__ vt) {
    __shared__ float t[32][33];
    const int bh = blockIdx.y >> 1, dblk = (blockIdx.y & 1) * 32;
    const int bb = bh >> 4, hh = bh & 15;
    const int key0 = blockIdx.x * 32;
    const int tx = threadIdx.x & 31, ty = threadIdx.x >> 5;
#pragma unroll
    for (int i = 0; i < 32; i += 8)
        t[ty + i][tx] = v[(size_t)(bb * SEQ_L + key0 + ty + i) * D_MODEL + hh * 64 + dblk + tx];
    __syncthreads();
#pragma unroll
    for (int i = 0; i < 32; i += 8)
        vt[((size_t)bh * 64 + dblk + ty + i) * SEQ_L + key0 + tx] = f2bf(t[tx][ty + i]);
}

// ---------------------------------------------------------------------------
// Fused GEMM: C[M,N] = A[M,K](bf16) @ Wt[N,K](bf16)^T + bias (+ epilogue).
// Block tile 128x64, 8 waves (4x2) of 32x32, K-step 32.
// Double-buffered LDS; 3 async b128 per thread per tile.
// ---------------------------------------------------------------------------
#define EP_BIAS  0   // + bias[n]
#define EP_WTE   1   // + bias[n] + wpe[(row % L)*N + n]
#define EP_RESID 2   // + bias[n] + extra[row*N + n]
#define EP_GELU  3   // gelu(acc + bias[n])

template<int MODE, int WF32, int WBF16>
__global__ __launch_bounds__(256)
void gemm_wmma_kernel(const __bf16* __restrict__ A, const __bf16* __restrict__ Bt,
                      const float* __restrict__ bias, const float* __restrict__ extra,
                      float* __restrict__ C, __bf16* __restrict__ Cb,
                      int N, int K, int Lmod) {
    __shared__ alignas(16) __bf16 As[2][128 * 32];   // [m][k], 8KB x2
    __shared__ alignas(16) __bf16 Bs[2][64 * 32];    // [n][k], 4KB x2

    const int tid  = threadIdx.x;
    const int lane = tid & 31;
    const int wave = tid >> 5;
    const int wm   = wave >> 1;           // 0..3  (M direction)
    const int wn   = wave & 1;            // 0..1  (N direction)
    const int m0   = blockIdx.y * 128;
    const int n0   = blockIdx.x * 64;
    const int ml   = lane & 15;
    const int kb   = (lane >> 4) * 8;

    // Stage one K-tile (async): A 512 x b128 (2/thread) + B 256 x b128 (1/thread).
    auto stage = [&](int k0, int buf) {
#pragma unroll
        for (int i = 0; i < 2; ++i) {
            int idx = tid + 256 * i;
            int r = idx >> 2, seg = idx & 3;
            cp_async16(&As[buf][r * 32 + seg * 8], &A[(size_t)(m0 + r) * K + k0 + seg * 8]);
        }
        int r = tid >> 2, seg = tid & 3;
        cp_async16(&Bs[buf][r * 32 + seg * 8], &Bt[(size_t)(n0 + r) * K + k0 + seg * 8]);
    };

    v8f zero = {0.f,0.f,0.f,0.f,0.f,0.f,0.f,0.f};
    v8f acc[2][2];
#pragma unroll
    for (int i = 0; i < 2; ++i)
#pragma unroll
        for (int j = 0; j < 2; ++j) acc[i][j] = zero;

    const int nsteps = K >> 5;
    stage(0, 0);
    for (int s = 0; s < nsteps; ++s) {
        const int buf = s & 1;
        if (s + 1 < nsteps) {
            stage((s + 1) << 5, buf ^ 1);   // overlap next tile's copies
            wait_async_n<3>();              // tile s landed; tile s+1 in flight
        } else {
            wait_async_n<0>();
        }
        __syncthreads();

        v16bf a0 = load_frag(&As[buf][(wm * 32 +  0 + ml) * 32], kb);
        v16bf a1 = load_frag(&As[buf][(wm * 32 + 16 + ml) * 32], kb);
        v16bf b0 = load_frag(&Bs[buf][(wn * 32 +  0 + ml) * 32], kb);
        v16bf b1 = load_frag(&Bs[buf][(wn * 32 + 16 + ml) * 32], kb);

        acc[0][0] = wmma_bf16(a0, b0, acc[0][0]);
        acc[0][1] = wmma_bf16(a0, b1, acc[0][1]);
        acc[1][0] = wmma_bf16(a1, b0, acc[1][0]);
        acc[1][1] = wmma_bf16(a1, b1, acc[1][1]);
        __syncthreads();   // buffer free before iteration s+1 restages it
    }

    const int hi = lane >> 4;
#pragma unroll
    for (int mg = 0; mg < 2; ++mg)
#pragma unroll
        for (int ng = 0; ng < 2; ++ng)
#pragma unroll
            for (int r = 0; r < 8; ++r) {
                int row = m0 + wm * 32 + mg * 16 + r + 8 * hi;
                int col = n0 + wn * 32 + ng * 16 + ml;
                float v = acc[mg][ng][r] + bias[col];
                if (MODE == EP_WTE)   v += extra[(size_t)(row % Lmod) * N + col];
                if (MODE == EP_RESID) v += extra[(size_t)row * N + col];
                if (MODE == EP_GELU)  v = 0.5f * v * (1.0f + erff(v * 0.70710678118654752f));
                if (WF32)  C [(size_t)row * N + col] = v;
                if (WBF16) Cb[(size_t)row * N + col] = f2bf(v);
            }
}

// ---------------------------------------------------------------------------
// LayerNorm: block per row; emits fp32 and bf16 outputs.
// ---------------------------------------------------------------------------
__global__ __launch_bounds__(256)
void ln_kernel(const float* __restrict__ x, const float* __restrict__ w,
               const float* __restrict__ b, float* __restrict__ y,
               __bf16* __restrict__ yb, int D) {
    __shared__ float red[256];
    const int row = blockIdx.x, tid = threadIdx.x;
    const float* xr = x + (size_t)row * D;
    float s = 0.f;
    for (int i = tid; i < D; i += 256) s += xr[i];
    red[tid] = s; __syncthreads();
    for (int off = 128; off > 0; off >>= 1) { if (tid < off) red[tid] += red[tid + off]; __syncthreads(); }
    float mu = red[0] / D; __syncthreads();
    float v2 = 0.f;
    for (int i = tid; i < D; i += 256) { float d = xr[i] - mu; v2 += d * d; }
    red[tid] = v2; __syncthreads();
    for (int off = 128; off > 0; off >>= 1) { if (tid < off) red[tid] += red[tid + off]; __syncthreads(); }
    float inv = rsqrtf(red[0] / D + 1e-5f);
    for (int i = tid; i < D; i += 256) {
        float o = (xr[i] - mu) * inv * w[i] + b[i];
        y [(size_t)row * D + i] = o;
        yb[(size_t)row * D + i] = f2bf(o);
    }
}

// ---------------------------------------------------------------------------
// Flash attention (causal). grid = (L/128, B*N_HEAD), block = 256 (8 waves).
// Double-buffered K/V tiles (2 async b128 per thread per tile).
// ---------------------------------------------------------------------------
__global__ __launch_bounds__(256)
void attn_kernel(const float* __restrict__ kq, const __bf16* __restrict__ kqb,
                 const __bf16* __restrict__ vt, __bf16* __restrict__ ob) {
    __shared__ alignas(16) __bf16 Ks[2][32 * 64];    // [key][d], 4KB x2
    __shared__ alignas(16) __bf16 Vs[2][64 * 32];    // [d][key], 4KB x2
    __shared__ alignas(16) __bf16 Ps[8][16 * 32];    // per-wave P tile [q][key]

    const int tid  = threadIdx.x;
    const int lane = tid & 31;
    const int wave = tid >> 5;
    const int ml   = lane & 15;
    const int hi   = lane >> 4;
    const int kb   = hi * 8;
    const int bh   = blockIdx.y;
    const int bb   = bh >> 4;
    const int hh   = bh & 15;
    const int qlo  = blockIdx.x * 128 + wave * 16;
    const int qhi_row = qlo + 15;
    const float scale = 0.125f;                       // 1/sqrt(64)

    auto stage = [&](int kb0, int buf) {
        {   // K tile: 32 rows x 128B = 256 x b128
            int key = tid >> 3, seg = tid & 7;
            cp_async16(&Ks[buf][key * 64 + seg * 8],
                       &kqb[(size_t)(bb * SEQ_L + kb0 + key) * (2 * D_MODEL) + hh * HEAD_DIM + seg * 8]);
        }
        {   // V tile: 64 rows x 64B = 256 x b128
            int d = tid >> 2, seg = tid & 3;
            cp_async16(&Vs[buf][d * 32 + seg * 8],
                       &vt[((size_t)bh * 64 + d) * SEQ_L + kb0 + seg * 8]);
        }
    };

    // Q fragments (fp32 source, scaled, converted once)
    const float* qrow = kq + (size_t)(bb * SEQ_L + qlo + ml) * (2 * D_MODEL) + D_MODEL + hh * HEAD_DIM;
    v16bf aq[2];
#pragma unroll
    for (int s = 0; s < 2; ++s)
#pragma unroll
        for (int j = 0; j < 8; ++j) {
            aq[s][j]     = f2bf(qrow[s * 32 + kb + j]      * scale);
            aq[s][8 + j] = f2bf(qrow[s * 32 + kb + 16 + j] * scale);
        }

    v8f zero = {0.f,0.f,0.f,0.f,0.f,0.f,0.f,0.f};
    v8f accO[4]; float m_i[8], l_i[8];
#pragma unroll
    for (int nf = 0; nf < 4; ++nf) accO[nf] = zero;
#pragma unroll
    for (int r = 0; r < 8; ++r) { m_i[r] = -INFINITY; l_i[r] = 0.f; }

    const int kend = blockIdx.x * 128 + 128;
    stage(0, 0);
    for (int kb0 = 0; kb0 < kend; kb0 += 32) {
        const int buf = (kb0 >> 5) & 1;
        if (kb0 + 32 < kend) {
            stage(kb0 + 32, buf ^ 1);
            wait_async_n<2>();
        } else {
            wait_async_n<0>();
        }
        __syncthreads();

        if (kb0 <= qhi_row) {   // wave-uniform: EXEC all-ones around WMMA
            v8f s0 = zero, s1 = zero;
#pragma unroll
            for (int s = 0; s < 2; ++s) {
                v16bf bk0 = load_frag(&Ks[buf][( 0 + ml) * 64 + s * 32], kb);
                v16bf bk1 = load_frag(&Ks[buf][(16 + ml) * 64 + s * 32], kb);
                s0 = wmma_bf16(aq[s], bk0, s0);
                s1 = wmma_bf16(aq[s], bk1, s1);
            }

            const int key0 = kb0 + ml, key1 = key0 + 16;
#pragma unroll
            for (int r = 0; r < 8; ++r) {
                int row = qlo + r + 8 * hi;
                float e0 = (key0 <= row) ? s0[r] : -INFINITY;
                float e1 = (key1 <= row) ? s1[r] : -INFINITY;
                float mb = fmaxf(e0, e1);
#pragma unroll
                for (int msk = 1; msk < 16; msk <<= 1) mb = fmaxf(mb, __shfl_xor(mb, msk, 32));
                float mn = fmaxf(m_i[r], mb);
                float p0 = __expf(e0 - mn);
                float p1 = __expf(e1 - mn);
                float rs = p0 + p1;
#pragma unroll
                for (int msk = 1; msk < 16; msk <<= 1) rs += __shfl_xor(rs, msk, 32);
                float alpha = __expf(m_i[r] - mn);
                l_i[r] = l_i[r] * alpha + rs;
                m_i[r] = mn;
#pragma unroll
                for (int nf = 0; nf < 4; ++nf) accO[nf][r] *= alpha;
                Ps[wave][(r + 8 * hi) * 32 + ml]      = f2bf(p0);
                Ps[wave][(r + 8 * hi) * 32 + ml + 16] = f2bf(p1);
            }

            v16bf ap = load_frag(&Ps[wave][ml * 32], kb);
#pragma unroll
            for (int nf = 0; nf < 4; ++nf) {
                v16bf bv = load_frag(&Vs[buf][(nf * 16 + ml) * 32], kb);
                accO[nf] = wmma_bf16(ap, bv, accO[nf]);
            }
        }
        __syncthreads();
    }

#pragma unroll
    for (int r = 0; r < 8; ++r) {
        float inv = 1.0f / l_i[r];
        int row = qlo + r + 8 * hi;
        size_t base = (size_t)(bb * SEQ_L + row) * D_MODEL + hh * HEAD_DIM;
#pragma unroll
        for (int nf = 0; nf < 4; ++nf)
            ob[base + nf * 16 + ml] = f2bf(accO[nf][r] * inv);
    }
}

// ---------------------------------------------------------------------------
extern "C" void kernel_launch(void* const* d_in, const int* in_sizes, int n_in,
                              void* d_out, int out_size, void* d_ws, size_t ws_size,
                              hipStream_t stream) {
    (void)in_sizes; (void)n_in; (void)out_size; (void)ws_size;
    const float* x      = (const float*)d_in[0];
    const float* wte_w  = (const float*)d_in[1];
    const float* wte_b  = (const float*)d_in[2];
    const float* wpe    = (const float*)d_in[3];
    const float* ln1_w  = (const float*)d_in[4];
    const float* ln1_b  = (const float*)d_in[5];
    const float* kq_w   = (const float*)d_in[6];
    const float* kq_b   = (const float*)d_in[7];
    const float* v_w    = (const float*)d_in[8];
    const float* v_b    = (const float*)d_in[9];
    const float* ao_w   = (const float*)d_in[10];
    const float* ao_b   = (const float*)d_in[11];
    const float* ln2_w  = (const float*)d_in[12];
    const float* ln2_b  = (const float*)d_in[13];
    const float* fc_w   = (const float*)d_in[14];
    const float* fc_b   = (const float*)d_in[15];
    const float* proj_w = (const float*)d_in[16];
    const float* proj_b = (const float*)d_in[17];

    float* out      = (float*)d_out;                          // [4096, 512]
    float* attn_out = out + (size_t)M_TOT * IN_DIM;           // [4096, 1024]

    // ---- workspace: fp32 region then bf16 region (all 16B aligned) ----
    float* fw  = (float*)d_ws;
    float* h   = fw;                                          //  4M f32
    float* hn  = h   + (size_t)M_TOT * D_MODEL;               //  4M f32
    float* kqf = hn  + (size_t)M_TOT * D_MODEL;               //  8M f32
    float* vf  = kqf + (size_t)M_TOT * 2 * D_MODEL;           //  4M f32
    float* xo  = vf  + (size_t)M_TOT * D_MODEL;               //  4M f32
    __bf16* bw    = (__bf16*)(xo + (size_t)M_TOT * D_MODEL);
    __bf16* xbf   = bw;                                       //  2M bf16
    __bf16* hnbf  = xbf   + (size_t)M_TOT * IN_DIM;           //  4M
    __bf16* kqbf  = hnbf  + (size_t)M_TOT * D_MODEL;          //  8M
    __bf16* vt    = kqbf  + (size_t)M_TOT * 2 * D_MODEL;      //  4M
    __bf16* obf   = vt    + (size_t)M_TOT * D_MODEL;          //  4M
    __bf16* xobf  = obf   + (size_t)M_TOT * D_MODEL;          //  4M
    __bf16* actbf = xobf  + (size_t)M_TOT * D_MODEL;          // 16M
    __bf16* wtewt = actbf + (size_t)M_TOT * MLP_HID;          // 0.5M
    __bf16* kqwt  = wtewt + (size_t)IN_DIM * D_MODEL;         //  2M
    __bf16* vwt   = kqwt  + (size_t)D_MODEL * 2 * D_MODEL;    //  1M
    __bf16* aowt  = vwt   + (size_t)D_MODEL * D_MODEL;        //  1M
    __bf16* fcwt  = aowt  + (size_t)D_MODEL * D_MODEL;        //  4M
    __bf16* pjwt  = fcwt  + (size_t)D_MODEL * MLP_HID;        //  2M

    const dim3 blk(256);
    const int MB = M_TOT / 128;   // 32 M-tiles

    // ---- prepass: convert activations/weights to bf16 (+ transpose W) ----
    cvt_bf16_kernel<<<(M_TOT * IN_DIM) / 256, blk, 0, stream>>>(x, xbf, M_TOT * IN_DIM);
    wtrans_kernel<<<dim3(D_MODEL / 32, IN_DIM / 32),      blk, 0, stream>>>(wte_w,  wtewt, IN_DIM,  D_MODEL);
    wtrans_kernel<<<dim3(2 * D_MODEL / 32, D_MODEL / 32), blk, 0, stream>>>(kq_w,   kqwt,  D_MODEL, 2 * D_MODEL);
    wtrans_kernel<<<dim3(D_MODEL / 32, D_MODEL / 32),     blk, 0, stream>>>(v_w,    vwt,   D_MODEL, D_MODEL);
    wtrans_kernel<<<dim3(D_MODEL / 32, D_MODEL / 32),     blk, 0, stream>>>(ao_w,   aowt,  D_MODEL, D_MODEL);
    wtrans_kernel<<<dim3(MLP_HID / 32, D_MODEL / 32),     blk, 0, stream>>>(fc_w,   fcwt,  D_MODEL, MLP_HID);
    wtrans_kernel<<<dim3(IN_DIM / 32, MLP_HID / 32),      blk, 0, stream>>>(proj_w, pjwt,  MLP_HID, IN_DIM);

    // 1) h = x @ wte_w + wte_b + wpe[row % L]
    gemm_wmma_kernel<EP_WTE, 1, 0><<<dim3(D_MODEL / 64, MB), blk, 0, stream>>>(
        xbf, wtewt, wte_b, wpe, h, nullptr, D_MODEL, IN_DIM, SEQ_L);
    // 2) hn = LN1(h)  (fp32 + bf16)
    ln_kernel<<<M_TOT, blk, 0, stream>>>(h, ln1_w, ln1_b, hn, hnbf, D_MODEL);
    // 3) kq = hn @ kq_w + kq_b  (fp32 for Q frags, bf16 for K tiles)
    gemm_wmma_kernel<EP_BIAS, 1, 1><<<dim3(2 * D_MODEL / 64, MB), blk, 0, stream>>>(
        hnbf, kqwt, kq_b, nullptr, kqf, kqbf, 2 * D_MODEL, D_MODEL, 0);
    // 4) v = hn @ v_w + v_b  (fp32, transposed next)
    gemm_wmma_kernel<EP_BIAS, 1, 0><<<dim3(D_MODEL / 64, MB), blk, 0, stream>>>(
        hnbf, vwt, v_b, nullptr, vf, nullptr, D_MODEL, D_MODEL, 0);
    // 5) Vt[b,h][d][key] bf16
    vtrans_kernel<<<dim3(SEQ_L / 32, BATCH * N_HEAD * 2), blk, 0, stream>>>(vf, vt);
    // 6) o = causal attention (bf16 out)
    attn_kernel<<<dim3(SEQ_L / 128, BATCH * N_HEAD), blk, 0, stream>>>(kqf, kqbf, vt, obf);
    // 7) attn_out = hn + (o @ ao_w + ao_b)   [second output]
    gemm_wmma_kernel<EP_RESID, 1, 0><<<dim3(D_MODEL / 64, MB), blk, 0, stream>>>(
        obf, aowt, ao_b, hn, attn_out, nullptr, D_MODEL, D_MODEL, 0);
    // 8) xo = LN2(attn_out)  (bf16 used downstream)
    ln_kernel<<<M_TOT, blk, 0, stream>>>(attn_out, ln2_w, ln2_b, xo, xobf, D_MODEL);
    // 9) act = gelu(xo @ fc_w + fc_b)  (bf16 only)
    gemm_wmma_kernel<EP_GELU, 0, 1><<<dim3(MLP_HID / 64, MB), blk, 0, stream>>>(
        xobf, fcwt, fc_b, nullptr, nullptr, actbf, MLP_HID, D_MODEL, 0);
    // 10) out = act @ proj_w + proj_b       [first output]
    gemm_wmma_kernel<EP_BIAS, 1, 0><<<dim3(IN_DIM / 64, MB), blk, 0, stream>>>(
        actbf, pjwt, proj_b, nullptr, out, nullptr, IN_DIM, MLP_HID, 0);
}
